// Pooling_24558622999100
// MI455X (gfx1250) — matrix operand order repaired and checked by reading, not compiled, not run on hardware
//
#include <hip/hip_runtime.h>
#include <hip/hip_bf16.h>

typedef __attribute__((ext_vector_type(16))) _Float16 v16h;
typedef __attribute__((ext_vector_type(8)))  float    v8f;

#define P_TOTAL     4096
#define PED_PER_BLK 16
#define THREADS     512   // 16 wave32 waves
#define OUT_CH      128

union H2 { uint32_t u; _Float16 h[2]; };

__device__ __forceinline__ uint32_t pack2(float x, float y) {
    H2 p; p.h[0] = (_Float16)x; p.h[1] = (_Float16)y; return p.u;
}

// Low 32 bits of a generic (flat) shared pointer == DS byte address
// (flat shared addresses are {shared_aperture, lds_offset[31:0]}).
__device__ __forceinline__ uint32_t lds_addr32(const void* p) {
    return (uint32_t)(uintptr_t)p;
}

extern "C" __global__ __launch_bounds__(THREADS)
void pool_embed_kernel(const float* __restrict__ obs1,
                       const float* __restrict__ obs2,
                       const float* __restrict__ W,
                       const float* __restrict__ bias,
                       float* __restrict__ out)
{
    // Dynamic LDS carve-up (total 140288 B):
    //   [0      ..  65536) : 16 fine grids, 32x32 cells, 2xf16 packed per cell
    //   [65536  ..  98304) : obs2 staged (float2[4096])   <- async copy
    //   [98304  .. 131072) : obs1 staged (float2[4096])   <- async copy
    //   [131072 .. 139264) : W as f16 [128][32]
    //   [139264 .. 140288) : coarse features f16 [16 peds][32]  (WMMA A staging)
    extern __shared__ char smem[];
    uint32_t* fine  = (uint32_t*)smem;
    float2*   sh_o2 = (float2*)(smem + 65536);
    float2*   sh_o1 = (float2*)(smem + 98304);
    _Float16* Wh    = (_Float16*)(smem + 131072);
    _Float16* cg    = (_Float16*)(smem + 139264);

    const int tid     = threadIdx.x;
    const int wave    = tid >> 5;
    const int lane    = tid & 31;
    const int pedbase = blockIdx.x * PED_PER_BLK;

    // ---------- Phase 0a: async-copy obs2/obs1 into LDS (CDNA5 ASYNC path) --
    // 32 KB per array, 512 lanes x 16 B per issue -> 4 issues per array.
    {
        const uint32_t lo2 = lds_addr32(sh_o2);
        const uint32_t lo1 = lds_addr32(sh_o1);
        #pragma unroll
        for (int k = 0; k < 4; ++k) {
            const uint32_t off = (uint32_t)tid * 16u + (uint32_t)k * 8192u;
            asm volatile("global_load_async_to_lds_b128 %0, %1, %2"
                         :: "v"(lo2 + off), "v"(off), "s"(obs2) : "memory");
            asm volatile("global_load_async_to_lds_b128 %0, %1, %2"
                         :: "v"(lo1 + off), "v"(off), "s"(obs1) : "memory");
        }
    }

    // ---------- Phase 0b: zero fine grids; stage f16 W (overlaps async) ----
    for (int k = tid; k < PED_PER_BLK * 1024; k += THREADS) fine[k] = 0u;
    for (int e = tid; e < OUT_CH * 32; e += THREADS) Wh[e] = (_Float16)W[e];

    asm volatile("s_wait_asynccnt 0x0" ::: "memory");
    __syncthreads();

    // ---------- Phase 1: all-pairs scatter into this wave's fine grid ------
    const int slot = wave;              // one pedestrian per wave
    const int i    = pedbase + slot;
    const float2 p2i = sh_o2[i];
    const float2 p1i = sh_o1[i];
    const float vix = p2i.x - p1i.x;
    const float viy = p2i.y - p1i.y;
    uint32_t* fg = fine + slot * 1024;
    for (int j0 = 0; j0 < P_TOTAL; j0 += 32) {
        const int j = j0 + lane;
        const float2 q2 = sh_o2[j];
        const float2 q1 = sh_o1[j];
        const float fx = (q2.x - p2i.x) * 4.0f + 16.0f;  // rel/(CELL/POOL)+G/2
        const float fy = (q2.y - p2i.y) * 4.0f + 16.0f;
        const bool valid = (j != i) && (q2.x == q2.x) &&
                           (fx >= 0.0f) && (fx < 32.0f) &&
                           (fy >= 0.0f) && (fy < 32.0f);
        if (valid) {
            const int cx = (int)fx, cy = (int)fy;        // floor (non-negative)
            const float vjx = q2.x - q1.x;
            const float vjy = q2.y - q1.y;
            fg[cx * 32 + cy] = pack2(vjx - vix, vjy - viy);   // "set" semantics
        }
    }
    __syncthreads();

    // ---------- Phase 2: 8x8 sum-pool fine -> coarse (f16 WMMA-A staging) --
    if (lane < 16) {
        const int c   = lane;          // coarse cell 0..15
        const int cxc = c >> 2, cyc = c & 3;
        const uint32_t* base = fine + slot * 1024 + (cxc * 8) * 32 + (cyc * 8);
        float sx = 0.0f, sy = 0.0f;
        #pragma unroll
        for (int px = 0; px < 8; ++px) {
            #pragma unroll
            for (int py = 0; py < 8; ++py) {
                H2 u; u.u = base[px * 32 + py];
                sx += (float)u.h[0];
                sy += (float)u.h[1];
            }
        }
        cg[slot * 32 + c]      = (_Float16)sx;  // pd=0 block: k = 0..15
        cg[slot * 32 + 16 + c] = (_Float16)sy;  // pd=1 block: k = 16..31
    }
    __syncthreads();

    // ---------- Phase 3: relu(cg[16x32] @ W^T + b) via one WMMA per tile ---
    if (wave < 8) {
        const int group = lane >> 4;        // 0: lanes 0-15, 1: lanes 16-31
        const int n     = lane & 15;        // tile column / A row
        const int o     = wave * 16 + n;    // output channel

        v16h a, bm;
        #pragma unroll
        for (int v = 0; v < 8; ++v) {
            // A (16x32 f16): lane=M%16; v0-3 hold K 0-7 / 8-15, v4-7 hold K 16-23 / 24-31
            const int ka = ((v < 4) ? 0 : 16) + group * 8 + 2 * (v & 3);
            a[2 * v]     = cg[n * 32 + ka];
            a[2 * v + 1] = cg[n * 32 + ka + 1];
            // B (32x16 f16): lane=N (per half); same K order as A: v0-3 hold K 0-7 / 8-15, v4-7 hold K 16-23 / 24-31
            const int kb = ((v < 4) ? 0 : 16) + group * 8 + 2 * (v & 3);
            bm[2 * v]     = Wh[o * 32 + kb];
            bm[2 * v + 1] = Wh[o * 32 + kb + 1];
        }

        const float bo = bias[o];
        v8f c;
        #pragma unroll
        for (int r = 0; r < 8; ++r) c[r] = bo;  // C[m][n] = bias[o]

        v8f d = __builtin_amdgcn_wmma_f32_16x16x32_f16(
            /*neg_a=*/false, a, /*neg_b=*/false, bm,
            /*c_mod=*/(short)0, c, /*reuse_a=*/false, /*reuse_b=*/false);

        #pragma unroll
        for (int r = 0; r < 8; ++r) {
            const int m = group * 8 + r;    // C/D layout: VGPR r -> M=r (+8 hi half)
            const float x = d[r];
            out[(size_t)(pedbase + m) * OUT_CH + o] = (x > 0.0f) ? x : 0.0f;
        }
    }
}

extern "C" void kernel_launch(void* const* d_in, const int* in_sizes, int n_in,
                              void* d_out, int out_size, void* d_ws, size_t ws_size,
                              hipStream_t stream) {
    (void)in_sizes; (void)n_in; (void)d_ws; (void)ws_size; (void)out_size;
    // setup_inputs order: hidden_state (unused), obs1, obs2, W, b
    const float* obs1 = (const float*)d_in[1];
    const float* obs2 = (const float*)d_in[2];
    const float* W    = (const float*)d_in[3];
    const float* bias = (const float*)d_in[4];
    float* out = (float*)d_out;

    const size_t shmem = 140288u;
    pool_embed_kernel<<<dim3(P_TOTAL / PED_PER_BLK), dim3(THREADS), shmem, stream>>>(
        obs1, obs2, W, bias, out);
}
